// SruEmb_46548855554156
// MI455X (gfx1250) — matrix operand order repaired and checked by reading, not compiled, not run on hardware
//
#include <hip/hip_runtime.h>
#include <hip/hip_bf16.h>
#include <math.h>

#define B_   32
#define T_   512
#define D_   1024
#define HOP_ 8
#define HID_ 350
#define HIDP 384            // HID padded to 3 * 128 for WMMA tiling
#define M_   (B_ * T_)      // 16384 tokens
#define N3D  (3 * D_)       // 3072
#define NEGF (-1e30f)

typedef _Float16 h16;
typedef h16  v16h __attribute__((ext_vector_type(16)));
typedef h16  v8h  __attribute__((ext_vector_type(8)));
typedef float v8f __attribute__((ext_vector_type(8)));

__device__ __forceinline__ v16h cat8(v8h lo, v8h hi) {
  v16h r;
#pragma unroll
  for (int i = 0; i < 8; ++i) { r[i] = lo[i]; r[8 + i] = hi[i]; }
  return r;
}

// CDNA5 async global->LDS copy (ASYNCcnt-tracked), GV addressing mode.
// LDS operand = low 32 bits of the generic shared pointer (group-segment offset).
__device__ __forceinline__ void async_load_b128(const h16* g, h16* l) {
  unsigned lds = (unsigned)(size_t)l;
  asm volatile("global_load_async_to_lds_b128 %0, %1, off"
               :: "v"(lds), "v"(g)
               : "memory");
}
__device__ __forceinline__ void wait_async() {
  asm volatile("s_wait_asynccnt 0x0" ::: "memory");
}

// ---------------------------------------------------------------------------
// f32 -> f16 conversion
// ---------------------------------------------------------------------------
__global__ void cvt_f16_kernel(const float* __restrict__ in, h16* __restrict__ out, int n) {
  int i = blockIdx.x * blockDim.x + threadIdx.x;
  if (i < n) out[i] = (h16)in[i];
}

// W (K=D_, N) f32 row-major  ->  Wt (N, K) f16  (transpose + convert)
__global__ void transW_kernel(const float* __restrict__ W, h16* __restrict__ Wt, int N) {
  int i = blockIdx.x * blockDim.x + threadIdx.x;   // index into Wt, k fastest
  if (i >= N * D_) return;
  int k = i % D_, n = i / D_;
  Wt[(size_t)n * D_ + k] = (h16)W[(size_t)k * N + n];
}

// ws1 (HID_, D_) f32 -> (HIDP, D_) f16 with zero padding rows
__global__ void cvt_ws1_kernel(const float* __restrict__ ws1, h16* __restrict__ Wt) {
  int i = blockIdx.x * blockDim.x + threadIdx.x;
  if (i >= HIDP * D_) return;
  int k = i % D_, n = i / D_;
  Wt[i] = (n < HID_) ? (h16)ws1[(size_t)n * D_ + k] : (h16)0.0f;
}

// ---------------------------------------------------------------------------
// WMMA GEMM:  C[M x N] (f32, ldc) = A[M x K] (f16, K-major) * Bt[N x K]^T
// Block: 256 thr / 8 waves. Tile 256(M) x 128(N), K-step 32, wave -> 64x64.
// Double-buffered LDS fed by global_load_async_to_lds_b128 (ASYNCcnt).
// ---------------------------------------------------------------------------
#define LDT 40   // LDS row stride in halves (32 + 8 pad, 16B aligned)
#define BM  256
#define BN  128

__global__ __launch_bounds__(256)
void gemm_wmma_kernel(const h16* __restrict__ A, const h16* __restrict__ Bt,
                      float* __restrict__ C, int K, int ldc, int do_tanh) {
  __shared__ __align__(32) h16 sA[2][BM * LDT];
  __shared__ __align__(32) h16 sB[2][BN * LDT];

  const int tid  = threadIdx.x;
  const int lane = tid & 31;
  const int wave = tid >> 5;
  const int wm   = wave & 3;   // 4 M sub-tiles of 64
  const int wn   = wave >> 2;  // 2 N sub-tiles of 64
  const int tileN = blockIdx.x * BN;
  const int tileM = blockIdx.y * BM;

  v8f acc[4][4];
#pragma unroll
  for (int mi = 0; mi < 4; ++mi)
#pragma unroll
    for (int ni = 0; ni < 4; ++ni)
#pragma unroll
      for (int e = 0; e < 8; ++e) acc[mi][ni][e] = 0.0f;

  // async tile feed: thread t -> A row t (4 chunks), B row t/2 (2 chunks)
  auto issue_tile = [&](int k0, int buf) {
    const h16* ga = A + (size_t)(tileM + tid) * K + k0;       // 64B contiguous
    h16* la = &sA[buf][tid * LDT];
    async_load_b128(ga,      la);
    async_load_b128(ga + 8,  la + 8);
    async_load_b128(ga + 16, la + 16);
    async_load_b128(ga + 24, la + 24);
    const h16* gb = Bt + (size_t)(tileN + (tid >> 1)) * K + k0 + (tid & 1) * 16;
    h16* lb = &sB[buf][(tid >> 1) * LDT + (tid & 1) * 16];
    async_load_b128(gb,     lb);
    async_load_b128(gb + 8, lb + 8);
  };

  const int lr   = lane & 15;  // lane row/col within 16
  const int lsel = lane >> 4;  // lane half-select

  issue_tile(0, 0);
  wait_async();
  __syncthreads();

  const int nsteps = K >> 5;
  int buf = 0;
  for (int s = 0; s < nsteps; ++s) {
    if (s + 1 < nsteps) issue_tile((s + 1) << 5, buf ^ 1);

    // ---- fragment loads per ISA 16-bit A/B VGPR layouts ----
    v16h afrag[4], bfrag[4];
#pragma unroll
    for (int mi = 0; mi < 4; ++mi) {
      const h16* ar = &sA[buf][(wm * 64 + mi * 16 + lr) * LDT];
      // lanes 0-15: K {0-7,16-23}; lanes 16-31: K {8-15,24-31}
      afrag[mi] = cat8(*(const v8h*)(ar + 8 * lsel),
                       *(const v8h*)(ar + 16 + 8 * lsel));
    }
#pragma unroll
    for (int ni = 0; ni < 4; ++ni) {
      const h16* br = &sB[buf][(wn * 64 + ni * 16 + lr) * LDT];
      // lanes 0-15: K {0-7,16-23} of col lr; lanes 16-31: K {8-15,24-31}
      bfrag[ni] = cat8(*(const v8h*)(br + 8 * lsel),
                       *(const v8h*)(br + 16 + 8 * lsel));
    }

#pragma unroll
    for (int mi = 0; mi < 4; ++mi)
#pragma unroll
      for (int ni = 0; ni < 4; ++ni)
        acc[mi][ni] = __builtin_amdgcn_wmma_f32_16x16x32_f16(
            false, afrag[mi], false, bfrag[ni],
            (short)0, acc[mi][ni], false, false);

    wait_async();      // next tile resident before it is published
    __syncthreads();
    buf ^= 1;
  }

  // ---- store C: VGPR i -> row (i + 8*lsel), col lr ----
#pragma unroll
  for (int mi = 0; mi < 4; ++mi)
#pragma unroll
    for (int ni = 0; ni < 4; ++ni)
#pragma unroll
      for (int i = 0; i < 8; ++i) {
        int row = tileM + wm * 64 + mi * 16 + lsel * 8 + i;
        int col = tileN + wn * 64 + ni * 16 + lr;
        float v = acc[mi][ni][i];
        if (do_tanh) v = tanhf(v);
        C[(size_t)row * ldc + col] = v;
      }
}

// ---------------------------------------------------------------------------
// SRU elementwise recurrence over T, one thread per (b, d) channel
// ---------------------------------------------------------------------------
__global__ void sru_scan_kernel(const float* __restrict__ U, const float* __restrict__ bias,
                                const float* __restrict__ xin, float* __restrict__ xout,
                                h16* __restrict__ xout16) {
  int idx = blockIdx.x * blockDim.x + threadIdx.x;
  if (idx >= B_ * D_) return;
  int b = idx / D_, d = idx % D_;
  const float bf = bias[d], br = bias[D_ + d];
  float c = 0.0f;
  for (int t = 0; t < T_; ++t) {
    size_t ub = (size_t)(b * T_ + t) * N3D;
    float xt = U[ub + d];
    float f  = 1.0f / (1.0f + __expf(-(U[ub + D_ + d] + bf)));
    float r  = 1.0f / (1.0f + __expf(-(U[ub + 2 * D_ + d] + br)));
    c = f * c + (1.0f - f) * xt;
    size_t xo = (size_t)(b * T_ + t) * D_ + d;
    float y = r * tanhf(c) + (1.0f - r) * xin[xo];
    xout[xo]   = y;
    xout16[xo] = (h16)y;
  }
}

// ---------------------------------------------------------------------------
// lengths[b] = T - count(input[b, :, 0] == 0)
// ---------------------------------------------------------------------------
__global__ __launch_bounds__(512)
void lengths_kernel(const float* __restrict__ input, int* __restrict__ len) {
  __shared__ int cnt;
  if (threadIdx.x == 0) cnt = 0;
  __syncthreads();
  int b = blockIdx.x, t = threadIdx.x;
  if (input[((size_t)b * T_ + t) * D_] == 0.0f) atomicAdd(&cnt, 1);
  __syncthreads();
  if (threadIdx.x == 0) len[b] = T_ - cnt;
}

// ---------------------------------------------------------------------------
// alphas[b,k,t] = sum_h hbar[b,t,h] * ws2[k,h]   (one block per token)
// ---------------------------------------------------------------------------
__global__ __launch_bounds__(128)
void alphas_kernel(const float* __restrict__ hbar, const float* __restrict__ ws2,
                   float* __restrict__ alphas) {
  __shared__ float hrow[HID_];
  int token = blockIdx.x;              // b*T + t
  for (int h = threadIdx.x; h < HID_; h += 128)
    hrow[h] = hbar[(size_t)token * HIDP + h];
  __syncthreads();
  int k = threadIdx.x;
  if (k < HOP_) {
    float s = 0.0f;
    for (int h = 0; h < HID_; ++h) s += hrow[h] * ws2[k * HID_ + h];
    int b = token / T_, t = token % T_;
    alphas[((size_t)b * HOP_ + k) * T_ + t] = s;
  }
}

// ---------------------------------------------------------------------------
// masked softmax over t for each (b, k) row; writes attn into d_out
// ---------------------------------------------------------------------------
__global__ __launch_bounds__(512)
void softmax_kernel(const float* __restrict__ alphas, const int* __restrict__ len,
                    float* __restrict__ attn) {
  __shared__ float red[512];
  int row = blockIdx.x;                // b*HOP + k
  int b = row / HOP_;
  int t = threadIdx.x;
  int L = len[b];
  float v = (t >= L) ? NEGF : alphas[(size_t)row * T_ + t];
  red[t] = v;
  __syncthreads();
  for (int s = 256; s > 0; s >>= 1) {
    if (t < s) red[t] = fmaxf(red[t], red[t + s]);
    __syncthreads();
  }
  float m = red[0];
  __syncthreads();
  float e = __expf(v - m);
  red[t] = e;
  __syncthreads();
  for (int s = 256; s > 0; s >>= 1) {
    if (t < s) red[t] += red[t + s];
    __syncthreads();
  }
  attn[(size_t)row * T_ + t] = e / red[0];
}

// ---------------------------------------------------------------------------
// pooled[b,k,d] = sum_t attn[b,k,t] * x[b,t,d]
// ---------------------------------------------------------------------------
__global__ void pooled_kernel(const float* __restrict__ attn, const float* __restrict__ x,
                              float* __restrict__ pooled) {
  int idx = blockIdx.x * blockDim.x + threadIdx.x;
  if (idx >= B_ * HOP_ * D_) return;
  int d = idx % D_;
  int k = (idx / D_) % HOP_;
  int b = idx / (D_ * HOP_);
  float s = 0.0f;
  for (int t = 0; t < T_; ++t)
    s += attn[((size_t)b * HOP_ + k) * T_ + t] * x[((size_t)b * T_ + t) * D_ + d];
  pooled[((size_t)b * HOP_ + k) * D_ + d] = s;
}

// ---------------------------------------------------------------------------
// out[b,j] = fc_b[j] + sum_i pooled[b,i] * fc_w[j,i]   (b fastest for broadcast)
// ---------------------------------------------------------------------------
__global__ void fc_kernel(const float* __restrict__ pooled, const float* __restrict__ fc_w,
                          const float* __restrict__ fc_b, float* __restrict__ out) {
  int idx = blockIdx.x * blockDim.x + threadIdx.x;
  if (idx >= B_ * D_) return;
  int b = idx & 31;
  int j = idx >> 5;
  float s = fc_b[j];
  const float* wr = fc_w + (size_t)j * (HOP_ * D_);
  const float* pr = pooled + (size_t)b * (HOP_ * D_);
  for (int i = 0; i < HOP_ * D_; ++i) s += pr[i] * wr[i];
  out[(size_t)b * D_ + j] = s;
}

// ---------------------------------------------------------------------------
extern "C" void kernel_launch(void* const* d_in, const int* in_sizes, int n_in,
                              void* d_out, int out_size, void* d_ws, size_t ws_size,
                              hipStream_t stream) {
  (void)in_sizes; (void)n_in; (void)out_size; (void)ws_size;

  const float* input = (const float*)d_in[0];
  // d_in[1] = lg (unused by reference math)
  const float* W0    = (const float*)d_in[2];
  const float* b0    = (const float*)d_in[3];
  const float* W1    = (const float*)d_in[4];
  const float* b1    = (const float*)d_in[5];
  const float* ws1   = (const float*)d_in[6];
  const float* ws2   = (const float*)d_in[7];
  const float* fc_w  = (const float*)d_in[8];
  const float* fc_b  = (const float*)d_in[9];

  float* out_p    = (float*)d_out;
  float* attn_out = out_p + (size_t)B_ * D_;   // (out, attn) concatenated

  // ---- workspace carve-up (byte offsets, 256-aligned) ----
  char* ws = (char*)d_ws;
  size_t off = 0;
  auto carve = [&](size_t bytes) { char* p = ws + off; off += (bytes + 255) & ~(size_t)255; return p; };
  float* U      = (float*)carve((size_t)M_ * N3D * sizeof(float));   // 192 MB
  h16*   x16    = (h16*)  carve((size_t)M_ * D_ * sizeof(h16));      // 32 MB
  h16*   Wt16   = (h16*)  carve((size_t)N3D * D_ * sizeof(h16));     // 6 MB
  float* x1     = (float*)carve((size_t)M_ * D_ * sizeof(float));    // 64 MB
  float* x2     = (float*)carve((size_t)M_ * D_ * sizeof(float));    // 64 MB
  float* alphas = (float*)carve((size_t)B_ * HOP_ * T_ * sizeof(float));
  float* pooled = (float*)carve((size_t)B_ * HOP_ * D_ * sizeof(float));
  int*   len    = (int*)  carve((size_t)B_ * sizeof(int));
  float* hbar   = U;   // U dead after scan2 -> reuse for (M_, HIDP) hbar

  // 1) sequence lengths
  lengths_kernel<<<B_, 512, 0, stream>>>(input, len);

  // 2) layer 1: convert, transpose weights, GEMM, scan
  cvt_f16_kernel<<<(M_ * D_ + 255) / 256, 256, 0, stream>>>(input, x16, M_ * D_);
  transW_kernel<<<(N3D * D_ + 255) / 256, 256, 0, stream>>>(W0, Wt16, N3D);
  gemm_wmma_kernel<<<dim3(N3D / BN, M_ / BM), 256, 0, stream>>>(x16, Wt16, U, D_, N3D, 0);
  sru_scan_kernel<<<(B_ * D_ + 255) / 256, 256, 0, stream>>>(U, b0, input, x1, x16);

  // 3) layer 2
  transW_kernel<<<(N3D * D_ + 255) / 256, 256, 0, stream>>>(W1, Wt16, N3D);
  gemm_wmma_kernel<<<dim3(N3D / BN, M_ / BM), 256, 0, stream>>>(x16, Wt16, U, D_, N3D, 0);
  sru_scan_kernel<<<(B_ * D_ + 255) / 256, 256, 0, stream>>>(U, b1, x1, x2, x16);

  // 4) attention: hbar = tanh(x2 @ ws1^T) via WMMA (N padded to 384, tanh fused)
  cvt_ws1_kernel<<<(HIDP * D_ + 255) / 256, 256, 0, stream>>>(ws1, Wt16);
  gemm_wmma_kernel<<<dim3(HIDP / BN, M_ / BM), 256, 0, stream>>>(x16, Wt16, hbar, D_, HIDP, 1);

  // 5) alphas, masked softmax (attn -> d_out), pooling, FC
  alphas_kernel<<<M_, 128, 0, stream>>>(hbar, ws2, alphas);
  softmax_kernel<<<B_ * HOP_, 512, 0, stream>>>(alphas, len, attn_out);
  pooled_kernel<<<(B_ * HOP_ * D_ + 255) / 256, 256, 0, stream>>>(attn_out, x2, pooled);
  fc_kernel<<<(B_ * D_ + 255) / 256, 256, 0, stream>>>(pooled, fc_w, fc_b, out_p);
}
